// EConly_85469849190489
// MI455X (gfx1250) — matrix operand order repaired and checked by reading, not compiled, not run on hardware
//
#include <hip/hip_runtime.h>

// ---------------------------------------------------------------------------
// Types / helpers
// ---------------------------------------------------------------------------
typedef __bf16 bf16_t;
typedef __attribute__((ext_vector_type(16))) __bf16 v16bf;
typedef __attribute__((ext_vector_type(8)))  __bf16 v8bf;
typedef __attribute__((ext_vector_type(8)))  float  v8f;
typedef __attribute__((ext_vector_type(4)))  float  v4f;

#define DEV __device__ __forceinline__

// Round-to-nearest-even fp32 -> bf16 bits (result in high 16 bits)
DEV unsigned bfbits(float f) {
  unsigned u = __builtin_bit_cast(unsigned, f);
  return u + 0x7fffu + ((u >> 16) & 1u);
}
// Pack two floats into one dword of bf16 pairs: lo -> [15:0], hi -> [31:16]
DEV unsigned pk2(float lo, float hi) {
  return (bfbits(lo) >> 16) | (bfbits(hi) & 0xffff0000u);
}
DEV unsigned long long pk4(v4f v) {
  unsigned d0 = pk2(v[0], v[1]);
  unsigned d1 = pk2(v[2], v[3]);
  return (unsigned long long)d0 | ((unsigned long long)d1 << 32);
}

DEV v8f zero8() { v8f z; for (int e = 0; e < 8; ++e) z[e] = 0.f; return z; }

DEV v8f wmma_bf16(v16bf a, v16bf b, v8f c) {
  return __builtin_amdgcn_wmma_f32_16x16x32_bf16(
      false, a, false, b, (short)0, c, false, false);
}

// A fragment (16x32) from a row-major LDS tile [16][stride] (stride mult of 8).
// ISA: lanes 0-15 hold K=0..7,16..23; lanes 16-31 hold K=8..15,24..31.
DEV v16bf frag_A(const bf16_t* tile, int stride, int lane) {
  int r  = lane & 15;
  int kb = (lane >> 4) << 3;
  const bf16_t* p = tile + r * stride + kb;
  v8bf lo = *(const v8bf*)(p);
  v8bf hi = *(const v8bf*)(p + 16);
  v16bf o;
  for (int e = 0; e < 8; ++e) { o[e] = lo[e]; o[e + 8] = hi[e]; }
  return o;
}

// B fragment (32x16) from LDS tile storing B^T: [n][k] row-major [16][stride].
// ISA: lanes 0-15 hold K=0..7,16..23; lanes 16-31 hold K=8..15,24..31 (col N = lane%16).
DEV v16bf frag_B(const bf16_t* tile, int stride, int lane) {
  int n  = lane & 15;
  int kb = (lane >> 4) << 3;
  const bf16_t* p = tile + n * stride + kb;
  v8bf lo = *(const v8bf*)(p);
  v8bf hi = *(const v8bf*)(p + 16);
  v16bf o;
  for (int e = 0; e < 8; ++e) { o[e] = lo[e]; o[e + 8] = hi[e]; }
  return o;
}

// ---------------------------------------------------------------------------
// GEMM: C[M,N] = act(A[M,K] @ W[K,N] + bias + resid)
// 256 threads (8 waves), 128x128 tile, wave tile 32x64 (2x4 WMMA subtiles),
// K stepped by 32. Requires K % 32 == 0, N % 4 == 0.
// ---------------------------------------------------------------------------
__global__ __launch_bounds__(256) void gemm_bf16_kernel(
    const float* __restrict__ A, int lda,
    const float* __restrict__ W, int ldw,
    const float* __restrict__ bias,
    const float* __restrict__ resid,
    float* __restrict__ C, int ldc,
    int M, int N, int K, int relu)
{
  __shared__ bf16_t As[128 * 40];                // [row][k], stride 40 halves
  __shared__ bf16_t Bs[128 * 40];                // [col][k] (W^T), stride 40

  const int tid  = threadIdx.x;
  const int lane = tid & 31;
  const int wave = tid >> 5;                     // 0..7
  const int m0 = blockIdx.y * 128;
  const int n0 = blockIdx.x * 128;
  const int wr = (wave >> 1) * 32;               // wave row offset (0/32/64/96)
  const int wc = (wave & 1) * 64;                // wave col offset (0/64)

  v8f acc[2][4];
  for (int i = 0; i < 2; ++i)
    for (int j = 0; j < 4; ++j) acc[i][j] = zero8();

  for (int k0 = 0; k0 < K; k0 += 32) {
    // Stage A tile 128x32: 1024 float4 slots, 4 per thread, packed b64 stores
    for (int it = 0; it < 4; ++it) {
      int idx = tid + it * 256;
      int row = idx >> 3;
      int c4  = (idx & 7) << 2;
      int gm  = m0 + row;
      v4f vv = {0.f, 0.f, 0.f, 0.f};
      if (gm < M) vv = *(const v4f*)(A + (size_t)gm * lda + k0 + c4);
      *(unsigned long long*)(As + row * 40 + c4) = pk4(vv);
    }
    // Stage B tile 32x128 transposed into [col][k]: pack k-pairs -> b32 stores
    for (int it = 0; it < 2; ++it) {
      int idx = tid + it * 256;                  // 0..511 (16 kpairs x 32 c4)
      int kp  = idx >> 5;                        // 0..15 -> k = 2*kp
      int c4  = (idx & 31) << 2;                 // 0..124
      int gn  = n0 + c4;
      v4f v0 = {0.f, 0.f, 0.f, 0.f};
      v4f v1 = {0.f, 0.f, 0.f, 0.f};
      if (gn < N) {
        v0 = *(const v4f*)(W + (size_t)(k0 + 2 * kp)     * ldw + gn);
        v1 = *(const v4f*)(W + (size_t)(k0 + 2 * kp + 1) * ldw + gn);
      }
      for (int j = 0; j < 4; ++j)
        *(unsigned*)(Bs + (c4 + j) * 40 + 2 * kp) = pk2(v0[j], v1[j]);
    }
    // Prefetch next K tiles into cache (global_prefetch_b8 path)
    if (k0 + 32 < K) {
      int prow = tid >> 3, pc4 = (tid & 7) << 2;
      if (m0 + prow < M)
        __builtin_prefetch(A + (size_t)(m0 + prow) * lda + k0 + 32 + pc4, 0, 1);
      int pk = tid >> 5, pn = (tid & 31) << 2;
      if (n0 + pn < N)
        __builtin_prefetch(W + (size_t)(k0 + 32 + pk) * ldw + n0 + pn, 0, 1);
    }
    __syncthreads();

    const bf16_t* Aw = As + wr * 40;
    const bf16_t* Bw = Bs + wc * 40;
    v16bf a0 = frag_A(Aw,           40, lane);
    v16bf a1 = frag_A(Aw + 16 * 40, 40, lane);
    v16bf b0 = frag_B(Bw,           40, lane);
    v16bf b1 = frag_B(Bw + 16 * 40, 40, lane);
    v16bf b2 = frag_B(Bw + 32 * 40, 40, lane);
    v16bf b3 = frag_B(Bw + 48 * 40, 40, lane);
    acc[0][0] = wmma_bf16(a0, b0, acc[0][0]);
    acc[0][1] = wmma_bf16(a0, b1, acc[0][1]);
    acc[0][2] = wmma_bf16(a0, b2, acc[0][2]);
    acc[0][3] = wmma_bf16(a0, b3, acc[0][3]);
    acc[1][0] = wmma_bf16(a1, b0, acc[1][0]);
    acc[1][1] = wmma_bf16(a1, b1, acc[1][1]);
    acc[1][2] = wmma_bf16(a1, b2, acc[1][2]);
    acc[1][3] = wmma_bf16(a1, b3, acc[1][3]);
    __syncthreads();
  }

  // Epilogue. C/D layout: VGPR e -> row e (lanes 0-15) / e+8 (lanes 16-31).
  const int rb = (lane < 16) ? 0 : 8;
  const int cb = lane & 15;
  for (int i = 0; i < 2; ++i)
    for (int j = 0; j < 4; ++j)
      for (int e = 0; e < 8; ++e) {
        int gm = m0 + wr + i * 16 + rb + e;
        int gn = n0 + wc + j * 16 + cb;
        if (gm < M && gn < N) {
          float val = acc[i][j][e];
          if (bias)  val += bias[gn];
          if (resid) val += resid[(size_t)gm * ldc + gn];
          if (relu)  val = fmaxf(val, 0.f);
          C[(size_t)gm * ldc + gn] = val;
        }
      }
}

// ---------------------------------------------------------------------------
// Fused attention for one (batch, head, branch, 32-query tile).
// grid = (16 qtiles, 4 heads, 32 batch), 64 threads (2 waves).
// ---------------------------------------------------------------------------
#define ATTN_SMEM_BYTES (32*512*4 + 512*4 + 2*(32*40*2) + 2*(32*520*2))

__global__ __launch_bounds__(64) void attn_kernel(
    const float* __restrict__ q,
    const float* __restrict__ k,
    const float* __restrict__ v,
    const float* __restrict__ adj,     // non-null for adj branch
    const float* __restrict__ dist,    // non-null for dist branch
    const float* __restrict__ mask,    // (B, S), 1 = padded
    float* __restrict__ out,
    int branch)
{
  extern __shared__ unsigned char smem_raw[];
  float*  Sc = (float*)smem_raw;                           // 32 x 512 f32
  float*  Ms = (float*)(smem_raw + 32 * 512 * 4);          // mask*NEG, 512 f32
  bf16_t* Qs = (bf16_t*)(smem_raw + 32 * 512 * 4 + 512 * 4);
  bf16_t* Ks = Qs + 32 * 40;
  bf16_t* Vs = Ks + 32 * 40;                               // V^T: 32 x 520
  bf16_t* Ps = Vs + 32 * 520;                              // probs: 32 x 520

  const int tid  = threadIdx.x;
  const int lane = tid & 31;
  const int wave = tid >> 5;
  const int q0 = blockIdx.x * 32;
  const int h  = blockIdx.y;
  const int b  = blockIdx.z;
  const float scale = 0.1767766952966369f;                 // 1/sqrt(32)

  // Stage mask*NEG (512 floats)
  for (int it = 0; it < 2; ++it) {
    int idx = tid + it * 64;                               // 0..127 float4 slots
    v4f mv = *(const v4f*)(mask + b * 512 + idx * 4);
    v4f ng = mv * (-1.0e9f);
    *(v4f*)(Ms + idx * 4) = ng;
  }
  // Stage Q (32x32) packed
  for (int it = 0; it < 4; ++it) {
    int idx = tid + it * 64;                               // 0..255
    int r = idx >> 3, c4 = (idx & 7) << 2;
    v4f vv = *(const v4f*)(q + ((size_t)(b * 512 + q0 + r)) * 128 + h * 32 + c4);
    *(unsigned long long*)(Qs + r * 40 + c4) = pk4(vv);
  }
  // Stage V^T (32 depth x 512 keys) packed along key pairs
  for (int it = 0; it < 4; ++it) {
    int kp = tid + it * 64;                                // 0..255 key pairs
    const float* vp0 = v + ((size_t)(b * 512 + 2 * kp))     * 128 + h * 32;
    const float* vp1 = v + ((size_t)(b * 512 + 2 * kp + 1)) * 128 + h * 32;
    for (int d4 = 0; d4 < 8; ++d4) {
      v4f a0 = *(const v4f*)(vp0 + d4 * 4);
      v4f a1 = *(const v4f*)(vp1 + d4 * 4);
      for (int j = 0; j < 4; ++j)
        *(unsigned*)(Vs + (d4 * 4 + j) * 520 + 2 * kp) = pk2(a0[j], a1[j]);
    }
  }
  // Prefetch the adjacency / dist rows that phase 2 will stream from HBM
  {
    const int r = tid >> 1, half = tid & 1;
    const size_t grow = ((size_t)b * 512 + q0 + r) * 512 + half * 256;
    const float* wsrc = dist ? dist : adj;
    for (int pp = 0; pp < 4; ++pp)
      __builtin_prefetch(wsrc + grow + pp * 64, 0, 1);
  }
  __syncthreads();

  // Phase 1: scores = (Q @ K^T) * scale   (DEPTH=32 -> one WMMA K-step)
  for (int kt = 0; kt < 16; ++kt) {
    for (int it = 0; it < 4; ++it) {
      int idx = tid + it * 64;
      int r = idx >> 3, c4 = (idx & 7) << 2;
      v4f vv = *(const v4f*)(k + ((size_t)(b * 512 + kt * 32 + r)) * 128 +
                             h * 32 + c4);
      *(unsigned long long*)(Ks + r * 40 + c4) = pk4(vv);
    }
    __syncthreads();
    v16bf a  = frag_A(Qs + wave * 16 * 40, 40, lane);
    v16bf b0 = frag_B(Ks,           40, lane);
    v16bf b1 = frag_B(Ks + 16 * 40, 40, lane);
    v8f s0 = wmma_bf16(a, b0, zero8());
    v8f s1 = wmma_bf16(a, b1, zero8());
    const int rb = (lane < 16) ? 0 : 8;
    const int cb = lane & 15;
    for (int e = 0; e < 8; ++e) {
      Sc[(wave * 16 + rb + e) * 512 + kt * 32 + cb]      = s0[e] * scale;
      Sc[(wave * 16 + rb + e) * 512 + kt * 32 + 16 + cb] = s1[e] * scale;
    }
    __syncthreads();
  }

  // Phase 2: masked softmax + weighting. Lane pair (tid, tid^1) shares a row.
  {
    const int r    = tid >> 1;
    const int half = tid & 1;
    const int cbeg = half * 256;
    float* srow = Sc + r * 512;
    const size_t grow = ((size_t)b * 512 + q0 + r) * 512;

    float m1 = -3.0e38f;
    for (int kk = cbeg; kk < cbeg + 256; kk += 4) {
      v4f sv = *(v4f*)(srow + kk);
      sv = sv + *(const v4f*)(Ms + kk);
      *(v4f*)(srow + kk) = sv;
      m1 = fmaxf(m1, fmaxf(fmaxf(sv[0], sv[1]), fmaxf(sv[2], sv[3])));
    }
    m1 = fmaxf(m1, __shfl_xor(m1, 1, 32));
    float Z1 = 0.f;
    for (int kk = cbeg; kk < cbeg + 256; kk += 4) {
      v4f sv = *(v4f*)(srow + kk);
      v4f ev;
      for (int j = 0; j < 4; ++j) { ev[j] = __expf(sv[j] - m1); Z1 += ev[j]; }
      *(v4f*)(srow + kk) = ev;
    }
    Z1 += __shfl_xor(Z1, 1, 32);

    float Z2 = 1.f;
    if (dist) {                                  // A *= softmax(dist + neg)
      float m2 = -3.0e38f;
      for (int kk = cbeg; kk < cbeg + 256; kk += 4) {
        v4f dv = *(const v4f*)(dist + grow + kk);
        dv = dv + *(const v4f*)(Ms + kk);
        m2 = fmaxf(m2, fmaxf(fmaxf(dv[0], dv[1]), fmaxf(dv[2], dv[3])));
      }
      m2 = fmaxf(m2, __shfl_xor(m2, 1, 32));
      Z2 = 0.f;
      for (int kk = cbeg; kk < cbeg + 256; kk += 4) {
        v4f dv = *(const v4f*)(dist + grow + kk);
        dv = dv + *(const v4f*)(Ms + kk);
        v4f sv = *(v4f*)(srow + kk);
        for (int j = 0; j < 4; ++j) {
          float e2 = __expf(dv[j] - m2);
          Z2 += e2;
          sv[j] *= e2;
        }
        *(v4f*)(srow + kk) = sv;
      }
      Z2 += __shfl_xor(Z2, 1, 32);
    }

    const float inv = 1.f / (Z1 * Z2);
    for (int kk = cbeg; kk < cbeg + 256; kk += 4) {
      v4f p = *(v4f*)(srow + kk);
      p = p * inv;
      if (adj) p = p * *(const v4f*)(adj + grow + kk);
      unsigned* dst = (unsigned*)(Ps + r * 520 + kk);
      dst[0] = pk2(p[0], p[1]);
      dst[1] = pk2(p[2], p[3]);
    }
  }
  __syncthreads();

  // Phase 3: out = P(32x512) @ V(512x32), 16 WMMA K-chunks
  v8f o0 = zero8(), o1 = zero8();
  for (int kc = 0; kc < 16; ++kc) {
    v16bf a  = frag_A(Ps + wave * 16 * 520 + kc * 32, 520, lane);
    v16bf b0 = frag_B(Vs + kc * 32,            520, lane);
    v16bf b1 = frag_B(Vs + 16 * 520 + kc * 32, 520, lane);
    o0 = wmma_bf16(a, b0, o0);
    o1 = wmma_bf16(a, b1, o1);
  }
  const int rb = (lane < 16) ? 0 : 8;
  const int cb = lane & 15;
  for (int e = 0; e < 8; ++e) {
    size_t row = (size_t)b * 512 + q0 + wave * 16 + rb + e;
    int c0 = branch * 128 + h * 32 + cb;
    out[row * 256 + c0]      = o0[e];
    out[row * 256 + c0 + 16] = o1[e];
  }
}

// ---------------------------------------------------------------------------
// y = LayerNorm(x + res) * g + b      (D = 256, one wave per row)
// ---------------------------------------------------------------------------
__global__ __launch_bounds__(32) void ln_kernel(
    const float* __restrict__ xin, const float* __restrict__ res,
    const float* __restrict__ g, const float* __restrict__ bt,
    float* __restrict__ y)
{
  const int row  = blockIdx.x;
  const int lane = threadIdx.x;
  float vals[8];
  const float* px = xin + (size_t)row * 256 + lane * 8;
  const float* pr = res + (size_t)row * 256 + lane * 8;
  float s = 0.f;
  for (int i = 0; i < 8; ++i) { vals[i] = px[i] + pr[i]; s += vals[i]; }
  for (int o = 16; o > 0; o >>= 1) s += __shfl_xor(s, o, 32);
  const float mean = s * (1.f / 256.f);
  float ss = 0.f;
  for (int i = 0; i < 8; ++i) { float d = vals[i] - mean; ss += d * d; }
  for (int o = 16; o > 0; o >>= 1) ss += __shfl_xor(ss, o, 32);
  const float inv = rsqrtf(ss * (1.f / 256.f) + 1e-6f);
  float* py = y + (size_t)row * 256 + lane * 8;
  const float* pg = g + lane * 8;
  const float* pb = bt + lane * 8;
  for (int i = 0; i < 8; ++i) py[i] = (vals[i] - mean) * inv * pg[i] + pb[i];
}

// pooled[b][d] = sum_s x[b,s,d] * (mask==0)
__global__ __launch_bounds__(64) void pool_kernel(
    const float* __restrict__ x, const float* __restrict__ mask,
    float* __restrict__ pooled)
{
  const int b = blockIdx.y;
  const int d = blockIdx.x * 64 + threadIdx.x;
  float s = 0.f;
  for (int ss = 0; ss < 512; ++ss)
    if (mask[b * 512 + ss] == 0.f)
      s += x[((size_t)(b * 512 + ss)) * 256 + d];
  pooled[b * 256 + d] = s;
}

// o = h * rsqrt(max(sum(h*h), 1e-12)) per row of 256
__global__ __launch_bounds__(32) void l2_kernel(
    const float* __restrict__ h, float* __restrict__ o)
{
  const int row  = blockIdx.x;
  const int lane = threadIdx.x;
  float vals[8];
  float ss = 0.f;
  const float* ph = h + (size_t)row * 256 + lane * 8;
  for (int i = 0; i < 8; ++i) { vals[i] = ph[i]; ss += vals[i] * vals[i]; }
  for (int m = 16; m > 0; m >>= 1) ss += __shfl_xor(ss, m, 32);
  const float inv = rsqrtf(fmaxf(ss, 1e-12f));
  float* po = o + (size_t)row * 256 + lane * 8;
  for (int i = 0; i < 8; ++i) po[i] = vals[i] * inv;
}

// ---------------------------------------------------------------------------
// Host orchestration
// ---------------------------------------------------------------------------
extern "C" void kernel_launch(void* const* d_in, const int* in_sizes, int n_in,
                              void* d_out, int out_size, void* d_ws, size_t ws_size,
                              hipStream_t stream)
{
  (void)in_sizes; (void)n_in; (void)out_size; (void)ws_size;
  const float* mol     = (const float*)d_in[0];
  const float* adj     = (const float*)d_in[1];
  const float* dist    = (const float*)d_in[2];
  const float* mask    = (const float*)d_in[3];
  const float* W_embed = (const float*)d_in[4];
  const float* b_embed = (const float*)d_in[5];
  const float* W_qkv   = (const float*)d_in[6];
  const float* b_qkv   = (const float*)d_in[7];
  const float* W_o     = (const float*)d_in[8];
  const float* b_o     = (const float*)d_in[9];
  const float* W_ff1   = (const float*)d_in[10];
  const float* b_ff1   = (const float*)d_in[11];
  const float* W_ff2   = (const float*)d_in[12];
  const float* b_ff2   = (const float*)d_in[13];
  const float* ln1_g   = (const float*)d_in[14];
  const float* ln1_b   = (const float*)d_in[15];
  const float* ln2_g   = (const float*)d_in[16];
  const float* ln2_b   = (const float*)d_in[17];
  const float* Wh1     = (const float*)d_in[18];
  const float* bh1     = (const float*)d_in[19];
  const float* Wh2     = (const float*)d_in[20];
  const float* bh2     = (const float*)d_in[21];
  float* out = (float*)d_out;

  const int M = 32 * 512;                // 16384 tokens
  float* ws = (float*)d_ws;
  size_t off = 0;
  float* x      = ws + off; off += (size_t)M * 256;
  float* x2     = ws + off; off += (size_t)M * 256;
  float* cat    = ws + off; off += (size_t)M * 256;
  float* o1     = ws + off; off += (size_t)M * 256;
  float* qkv    = ws + off; off += (size_t)6 * M * 128;
  float* ff     = ws + off; off += (size_t)M * 1024;
  float* pooled = ws + off; off += 32 * 256;
  float* h1     = ws + off; off += 32 * 512;
  float* h2     = ws + off; off += 32 * 256;

  auto gemm = [&](const float* A, int lda, const float* W, int ldw,
                  const float* bias, const float* resid,
                  float* Cp, int ldc, int Mm, int Nn, int Kk, int relu) {
    dim3 g((Nn + 127) / 128, (Mm + 127) / 128);
    gemm_bf16_kernel<<<g, 256, 0, stream>>>(A, lda, W, ldw, bias, resid,
                                            Cp, ldc, Mm, Nn, Kk, relu);
  };

  // Embedding: (16384 x 64) @ (64 x 256)
  gemm(mol, 64, W_embed, 256, b_embed, nullptr, x, 256, M, 256, 64, 0);

  for (int i = 0; i < 4; ++i) {
    // 6 QKV projections (dist: q,k,v = 0,1,2 ; adj: 3,4,5)
    for (int j = 0; j < 6; ++j)
      gemm(x, 256, W_qkv + (size_t)(i * 6 + j) * 256 * 128, 128,
           b_qkv + (i * 6 + j) * 128, nullptr,
           qkv + (size_t)j * M * 128, 128, M, 128, 256, 0);

    dim3 ag(16, 4, 32);
    attn_kernel<<<ag, 64, ATTN_SMEM_BYTES, stream>>>(
        qkv, qkv + (size_t)1 * M * 128, qkv + (size_t)2 * M * 128,
        nullptr, dist, mask, cat, /*branch=*/0);
    attn_kernel<<<ag, 64, ATTN_SMEM_BYTES, stream>>>(
        qkv + (size_t)3 * M * 128, qkv + (size_t)4 * M * 128,
        qkv + (size_t)5 * M * 128, adj, nullptr, mask, cat, /*branch=*/1);

    // Output projection + residual LN
    gemm(cat, 256, W_o + (size_t)i * 256 * 256, 256, b_o + i * 256,
         nullptr, o1, 256, M, 256, 256, 0);
    ln_kernel<<<M, 32, 0, stream>>>(x, o1, ln1_g + i * 256, ln1_b + i * 256, x2);

    // FFN + residual LN
    gemm(x2, 256, W_ff1 + (size_t)i * 256 * 1024, 1024, b_ff1 + i * 1024,
         nullptr, ff, 1024, M, 1024, 256, 1);
    gemm(ff, 1024, W_ff2 + (size_t)i * 1024 * 256, 256, b_ff2 + i * 256,
         nullptr, o1, 256, M, 256, 1024, 0);
    ln_kernel<<<M, 32, 0, stream>>>(x2, o1, ln2_g + i * 256, ln2_b + i * 256, x);
  }

  // Masked pooling
  pool_kernel<<<dim3(4, 32), 64, 0, stream>>>(x, mask, pooled);

  // 4 contrastive heads: relu(relu(pooled@Wh1+b)@Wh2+b) then L2-normalize
  for (int c = 0; c < 4; ++c) {
    gemm(pooled, 256, Wh1 + (size_t)c * 256 * 512, 512, bh1 + c * 512,
         nullptr, h1, 512, 32, 512, 256, 1);
    gemm(h1, 512, Wh2 + (size_t)c * 512 * 256, 256, bh2 + c * 256,
         nullptr, h2, 256, 32, 256, 512, 1);
    l2_kernel<<<32, 32, 0, stream>>>(h2, out + (size_t)c * 32 * 256);
  }
}
